// LSTM_NER_TeacherForcing_9689446220596
// MI455X (gfx1250) — compile-verified, matrix operand order checked by reading, not run on hardware
//
#include <hip/hip_runtime.h>
#include <hip/hip_bf16.h>

// ---------------------------------------------------------------------------
// LSTM NER (B=32,T=512,E=512,H=512,V=50000,C=32,L=2) for MI455X / gfx1250.
//   - f16 storage, f32 WMMA accumulate (v_wmma_f32_16x16x32_f16).
//   - Input GEMMs: block-tiled 128(M)x64(N); W slab staged in LDS via
//     GLOBAL_LOAD_ASYNC_TO_LDS_B128 (ASYNCcnt pipelined), shared by 8 waves.
//   - Recurrent scan: one WG per (16-batch slice, direction); wave w owns
//     hidden cols [16w,16w+16) whose 4 gate tiles are WMMA accumulators, so
//     the LSTM cell update is elementwise on v8f regs; c lives in VGPRs for
//     all 512 steps; h double-buffered in LDS (1 s_barrier/step); Whh k-slices
//     async-staged per wave into dynamic LDS one slice ahead.
//   - Classifier+log_softmax: 1 wave per (b,t), lane=class, wave32 shuffles.
// ---------------------------------------------------------------------------

typedef __attribute__((ext_vector_type(16))) _Float16 v16h;
typedef __attribute__((ext_vector_type(8)))  float    v8f;
typedef __attribute__((ext_vector_type(8)))  _Float16 v8h;
typedef __attribute__((ext_vector_type(2)))  _Float16 v2h;
typedef __attribute__((ext_vector_type(4)))  int      v4i;

#define WMMA_F16(a, b, c) \
  __builtin_amdgcn_wmma_f32_16x16x32_f16(false, (a), false, (b), (short)0, (c), false, false)

// ---- CDNA5 async global->LDS copy (ASYNCcnt), with safe fallback -----------
#if defined(__has_builtin)
#if __has_builtin(__builtin_amdgcn_global_load_async_to_lds_b128)
#define HAVE_ASYNC_LDS 1
#endif
#endif
#ifndef HAVE_ASYNC_LDS
#define HAVE_ASYNC_LDS 0
#endif

#if HAVE_ASYNC_LDS
// signature (from clang diagnostic): (AS1 v4i*, AS3 v4i*, imm int, imm int)
#define ASYNC_COPY16(l, g)                                          \
  __builtin_amdgcn_global_load_async_to_lds_b128(                   \
      (__attribute__((address_space(1))) v4i*)(g),                  \
      (__attribute__((address_space(3))) v4i*)(l), 0, 0)
#if __has_builtin(__builtin_amdgcn_s_wait_asynccnt)
#define ASYNC_WAIT1() __builtin_amdgcn_s_wait_asynccnt(1)
#define ASYNC_WAIT8() __builtin_amdgcn_s_wait_asynccnt(8)
#else
#define ASYNC_WAIT1() asm volatile("s_wait_asynccnt 0x1" ::: "memory")
#define ASYNC_WAIT8() asm volatile("s_wait_asynccnt 0x8" ::: "memory")
#endif
#else
#define ASYNC_COPY16(l, g) (*(v8h*)(l) = *(const v8h*)(g))
#define ASYNC_WAIT1() ((void)0)
#define ASYNC_WAIT8() ((void)0)
#endif

constexpr int Bsz = 32, Tsz = 512, Esz = 512, Hsz = 512, Csz = 32;
constexpr int G4H = 4 * Hsz;  // 2048 gate dim

// 16x32 f16 fragment load (A: lane=row, B: lane=col) from row-major storage.
// CDNA5 16-bit operand layout: lane halves differ by K+8; VGPRs 0-3 hold
// K 0..7, VGPRs 4-7 hold K 16..23 (per half), dword-paired -> 32-bit loads.
__device__ inline v16h load_frag(const _Float16* __restrict__ base, int ldk,
                                 int row0, int k0, int lane) {
  const _Float16* p =
      base + (size_t)(row0 + (lane & 15)) * (size_t)ldk + k0 + ((lane >> 4) << 3);
  v16h out;
#pragma unroll
  for (int pp = 0; pp < 8; ++pp) {
    const int off = (pp < 4) ? (2 * pp) : (16 + 2 * (pp - 4));
    v2h pr = *(const v2h*)(p + off);
    out[2 * pp]     = pr[0];
    out[2 * pp + 1] = pr[1];
  }
  return out;
}

__device__ inline float sigmoidf_(float x) { return 1.0f / (1.0f + __expf(-x)); }

// ---------------------------------------------------------------- embed gather
__global__ void embed_kernel(const int* __restrict__ x, const float* __restrict__ emb,
                             _Float16* __restrict__ out) {
  size_t i = (size_t)blockIdx.x * blockDim.x + threadIdx.x;
  if (i >= (size_t)Bsz * Tsz * Esz) return;
  size_t row = i / Esz;
  int col = (int)(i % Esz);
  out[i] = (_Float16)emb[(size_t)x[row] * Esz + col];
}

// ---------------------------------------------------------------- f32 -> f16
__global__ void cvt_f16_kernel(const float* __restrict__ in, _Float16* __restrict__ out,
                               int n) {
  int i = blockIdx.x * blockDim.x + threadIdx.x;
  if (i < n) out[i] = (_Float16)in[i];
}

// ------------------------------------------------- input GEMM: xw = A@W^T + b
// Block: 8 waves, tile 128(M) x 64(N). W slab (64 rows x 32 K) async-staged
// into double-buffered LDS shared by all waves; A frags straight from global.
__global__ __launch_bounds__(256) void input_gemm_kernel(
    const _Float16* __restrict__ A, const _Float16* __restrict__ W,
    const float* __restrict__ bih, const float* __restrict__ bhh,
    _Float16* __restrict__ xw, int K) {
  __shared__ _Float16 wbuf[2][64][32];  // 8KB double buffer
  const int lane = threadIdx.x & 31;
  const int wave = threadIdx.x >> 5;  // M tile within block
  const int NS = G4H / 64;            // 32 N slabs
  const int n0 = (blockIdx.x % NS) * 64;
  const int m0 = (blockIdx.x / NS) * 128 + wave * 16;
  const int nk = K / 32;

  // each of 256 threads owns one 16B chunk of the 64x32-half slab
  const int srow = threadIdx.x >> 2;
  const int spart = (threadIdx.x & 3) * 8;
  const _Float16* wsrc = W + (size_t)(n0 + srow) * K + spart;

  v8f acc[4] = {};
  ASYNC_COPY16(&wbuf[0][srow][spart], wsrc);  // stage slice 0
  for (int it = 0; it < nk; ++it) {
    const int cb = it & 1;
    const int knext = (it + 1 < nk) ? (it + 1) * 32 : 0;  // harmless wrap
    ASYNC_COPY16(&wbuf[cb ^ 1][srow][spart], wsrc + knext);
    ASYNC_WAIT1();    // slice `it` (this wave's chunk) retired
    __syncthreads();  // -> visible to all consumer waves
    v16h a = load_frag(A, K, m0, it * 32, lane);
#pragma unroll
    for (int q = 0; q < 4; ++q) {
      v16h b = load_frag(&wbuf[cb][0][0], 32, 16 * q, 0, lane);
      acc[q] = WMMA_F16(a, b, acc[q]);
    }
    __syncthreads();  // all reads of cb done before it is overwritten
  }

  const int nlo = lane & 15, half8 = (lane >> 4) << 3;
#pragma unroll
  for (int q = 0; q < 4; ++q) {
    const int col = n0 + 16 * q + nlo;
    const float bias = bih[col] + bhh[col];
#pragma unroll
    for (int r = 0; r < 8; ++r)
      xw[(size_t)(m0 + r + half8) * G4H + col] = (_Float16)(acc[q][r] + bias);
  }
}

// --------------------------------------------------------- recurrent LSTM scan
// grid = (2 batch slices, 2 directions), block = 1024 threads (32 waves).
// Dynamic LDS: per-wave double-buffered Whh slice (64 gate-rows x 32 K) = 256KB.
__global__ __launch_bounds__(1024) void lstm_scan_kernel(
    const _Float16* __restrict__ xw_f, const _Float16* __restrict__ xw_b,
    const _Float16* __restrict__ Whh_f, const _Float16* __restrict__ Whh_b,
    _Float16* __restrict__ outH) {
  __shared__ _Float16 hbuf[2][16][Hsz];  // 32KB double-buffered h state
  extern __shared__ _Float16 dynsmem[];  // 32 waves * 2 * 64 * 32 halves

  const int dir = blockIdx.y;
  const int b0  = blockIdx.x * 16;
  const _Float16* __restrict__ xw  = dir ? xw_b : xw_f;
  const _Float16* __restrict__ Whh = dir ? Whh_b : Whh_f;

  const int wave  = threadIdx.x >> 5;
  const int lane  = threadIdx.x & 31;
  const int j0    = wave * 16;  // hidden column tile owned by this wave
  const int nlo   = lane & 15;
  const int half8 = (lane >> 4) << 3;
  _Float16* wv = dynsmem + (size_t)wave * (2 * 64 * 32);  // private region

  for (int i = threadIdx.x; i < 16 * Hsz; i += blockDim.x)
    (&hbuf[0][0][0])[i] = (_Float16)0.0f;
  __syncthreads();

  // per-wave async stage of one 64x32 Whh k-slice (8 x 16B chunks per lane)
  auto stageS = [&](int kt, int buf) {
#pragma unroll
    for (int i = 0; i < 8; ++i) {
      const int c = (i << 5) + lane;  // 0..255 chunk id
      const int row = c >> 2;         // 0..63 = gate*16 + nloc
      const int part = (c & 3) * 8;
      const _Float16* g =
          Whh + (size_t)((row >> 4) * Hsz + j0 + (row & 15)) * Hsz + kt + part;
      ASYNC_COPY16(wv + buf * 2048 + row * 32 + part, g);
    }
  };

  v8f c = {};       // cell state tile, resident in VGPRs for all 512 steps
  stageS(0, 0);     // prime the pipeline
  for (int step = 0; step < Tsz; ++step) {
    const int t   = dir ? (Tsz - 1 - step) : step;
    const int cur = step & 1, nxt = cur ^ 1;

    v8f gi, gf, gg, go;  // init gates from xw (biases folded)
#pragma unroll
    for (int r = 0; r < 8; ++r) {
      const _Float16* xr = xw + ((size_t)(b0 + r + half8) * Tsz + t) * (size_t)G4H;
      gi[r] = (float)xr[0 * Hsz + j0 + nlo];
      gf[r] = (float)xr[1 * Hsz + j0 + nlo];
      gg[r] = (float)xr[2 * Hsz + j0 + nlo];
      go[r] = (float)xr[3 * Hsz + j0 + nlo];
    }

    const _Float16* hcur = &hbuf[cur][0][0];
    for (int it = 0; it < Hsz / 32; ++it) {
      const int cb = it & 1;
      const int knext = (it + 1 < Hsz / 32) ? (it + 1) * 32 : 0;  // wraps to
      stageS(knext, cb ^ 1);  // next step's slice 0 at the last iteration
      ASYNC_WAIT8();          // previous slice retired (per-wave counter)
      v16h a = load_frag(hcur, Hsz, 0, it * 32, lane);
      const _Float16* wb = wv + cb * 2048;
      gi = WMMA_F16(a, load_frag(wb, 32, 0, 0, lane), gi);
      gf = WMMA_F16(a, load_frag(wb, 32, 16, 0, lane), gf);
      gg = WMMA_F16(a, load_frag(wb, 32, 32, 0, lane), gg);
      go = WMMA_F16(a, load_frag(wb, 32, 48, 0, lane), go);
    }

    // LSTM cell update: elementwise on matching C-layout elements
#pragma unroll
    for (int r = 0; r < 8; ++r) {
      float cn = sigmoidf_(gf[r]) * c[r] + sigmoidf_(gi[r]) * tanhf(gg[r]);
      float hn = sigmoidf_(go[r]) * tanhf(cn);
      c[r] = cn;
      _Float16 hh = (_Float16)hn;
      hbuf[nxt][r + half8][j0 + nlo] = hh;
      outH[((size_t)(b0 + r + half8) * Tsz + t) * (size_t)(2 * Hsz) + dir * Hsz + j0 + nlo] = hh;
    }
    __syncthreads();
  }
}

// -------------------------------------------- classifier + log_softmax (B,C,T)
__global__ __launch_bounds__(256) void classifier_kernel(
    const _Float16* __restrict__ h2, const float* __restrict__ truth,
    const int* __restrict__ tf, const float* __restrict__ Wc,
    const float* __restrict__ bc, float* __restrict__ out) {
  const int waveG = (int)((blockIdx.x * blockDim.x + threadIdx.x) >> 5);  // (b,t)
  const int lane  = threadIdx.x & 31;                                     // class
  const int b = waveG / Tsz, t = waveG % Tsz;

  const float* wrow = Wc + (size_t)lane * (2 * Hsz + 1);
  const float past = (tf[0] != 0 && t > 0) ? truth[(size_t)b * Tsz + (t - 1)] : 0.0f;
  float acc = wrow[0] * past + bc[lane];
  const _Float16* hrow = h2 + ((size_t)b * Tsz + t) * (size_t)(2 * Hsz);
#pragma unroll 8
  for (int j = 0; j < 2 * Hsz; ++j) acc += wrow[1 + j] * (float)hrow[j];

  float mx = acc;
#pragma unroll
  for (int o = 16; o; o >>= 1) mx = fmaxf(mx, __shfl_xor(mx, o, 32));
  float e = __expf(acc - mx), s = e;
#pragma unroll
  for (int o = 16; o; o >>= 1) s += __shfl_xor(s, o, 32);
  out[((size_t)b * Csz + lane) * Tsz + t] = acc - mx - __logf(s);
}

// ---------------------------------------------------------------------------
extern "C" void kernel_launch(void* const* d_in, const int* in_sizes, int n_in,
                              void* d_out, int out_size, void* d_ws, size_t ws_size,
                              hipStream_t stream) {
  (void)in_sizes; (void)n_in; (void)out_size; (void)ws_size;

  const int*   x     = (const int*)d_in[0];
  const float* truth = (const float*)d_in[1];
  const int*   tf    = (const int*)d_in[2];
  const float* emb   = (const float*)d_in[3];
  const float *Wih0f = (const float*)d_in[4],  *Whh0f = (const float*)d_in[5];
  const float *bih0f = (const float*)d_in[6],  *bhh0f = (const float*)d_in[7];
  const float *Wih0b = (const float*)d_in[8],  *Whh0b = (const float*)d_in[9];
  const float *bih0b = (const float*)d_in[10], *bhh0b = (const float*)d_in[11];
  const float *Wih1f = (const float*)d_in[12], *Whh1f = (const float*)d_in[13];
  const float *bih1f = (const float*)d_in[14], *bhh1f = (const float*)d_in[15];
  const float *Wih1b = (const float*)d_in[16], *Whh1b = (const float*)d_in[17];
  const float *bih1b = (const float*)d_in[18], *bhh1b = (const float*)d_in[19];
  const float *Wc    = (const float*)d_in[20], *bc    = (const float*)d_in[21];
  float* out = (float*)d_out;

  char*  ws  = (char*)d_ws;
  size_t off = 0;
  auto carve = [&](size_t bytes) {
    void* p = ws + off;
    off += (bytes + 255) & ~(size_t)255;
    return p;
  };
  _Float16* xemb = (_Float16*)carve((size_t)Bsz * Tsz * Esz * 2);
  _Float16* h1   = (_Float16*)carve((size_t)Bsz * Tsz * 2 * Hsz * 2);
  _Float16* h2   = (_Float16*)carve((size_t)Bsz * Tsz * 2 * Hsz * 2);
  _Float16* xwF  = (_Float16*)carve((size_t)Bsz * Tsz * G4H * 2);
  _Float16* xwB  = (_Float16*)carve((size_t)Bsz * Tsz * G4H * 2);
  _Float16* wIh0f = (_Float16*)carve((size_t)G4H * Esz * 2);
  _Float16* wIh0b = (_Float16*)carve((size_t)G4H * Esz * 2);
  _Float16* wHh0f = (_Float16*)carve((size_t)G4H * Hsz * 2);
  _Float16* wHh0b = (_Float16*)carve((size_t)G4H * Hsz * 2);
  _Float16* wIh1f = (_Float16*)carve((size_t)G4H * 2 * Hsz * 2);
  _Float16* wIh1b = (_Float16*)carve((size_t)G4H * 2 * Hsz * 2);
  _Float16* wHh1f = (_Float16*)carve((size_t)G4H * Hsz * 2);
  _Float16* wHh1b = (_Float16*)carve((size_t)G4H * Hsz * 2);

  {
    size_t n = (size_t)Bsz * Tsz * Esz;
    embed_kernel<<<(unsigned)((n + 255) / 256), 256, 0, stream>>>(x, emb, xemb);
  }
  auto cvt = [&](const float* src, _Float16* dst, int n) {
    cvt_f16_kernel<<<(n + 255) / 256, 256, 0, stream>>>(src, dst, n);
  };
  cvt(Wih0f, wIh0f, G4H * Esz);     cvt(Wih0b, wIh0b, G4H * Esz);
  cvt(Whh0f, wHh0f, G4H * Hsz);     cvt(Whh0b, wHh0b, G4H * Hsz);
  cvt(Wih1f, wIh1f, G4H * 2 * Hsz); cvt(Wih1b, wIh1b, G4H * 2 * Hsz);
  cvt(Whh1f, wHh1f, G4H * Hsz);     cvt(Whh1b, wHh1b, G4H * Hsz);

  // 256KB dynamic LDS for the scan kernel's Whh staging (WGP has 320KB)
  const int scanDynLds = 32 * 2 * 64 * 32 * (int)sizeof(_Float16);
  (void)hipFuncSetAttribute((const void*)lstm_scan_kernel,
                            hipFuncAttributeMaxDynamicSharedMemorySize, scanDynLds);

  const int gemmBlocks = (Bsz * Tsz / 128) * (G4H / 64);  // 4096
  input_gemm_kernel<<<gemmBlocks, 256, 0, stream>>>(xemb, wIh0f, bih0f, bhh0f, xwF, Esz);
  input_gemm_kernel<<<gemmBlocks, 256, 0, stream>>>(xemb, wIh0b, bih0b, bhh0b, xwB, Esz);
  lstm_scan_kernel<<<dim3(2, 2), 1024, scanDynLds, stream>>>(xwF, xwB, wHh0f, wHh0b, h1);

  input_gemm_kernel<<<gemmBlocks, 256, 0, stream>>>(h1, wIh1f, bih1f, bhh1f, xwF, 2 * Hsz);
  input_gemm_kernel<<<gemmBlocks, 256, 0, stream>>>(h1, wIh1b, bih1b, bhh1b, xwB, 2 * Hsz);
  lstm_scan_kernel<<<dim3(2, 2), 1024, scanDynLds, stream>>>(xwF, xwB, wHh1f, wHh1b, h2);

  classifier_kernel<<<(Bsz * Tsz) / 8, 256, 0, stream>>>(h2, truth, tf, Wc, bc, out);
}
